// RecurrentBlock_50483045597231
// MI455X (gfx1250) — matrix operand order repaired and checked by reading, not compiled, not run on hardware
//
#include <hip/hip_runtime.h>
#include <stdint.h>

// ---------------------------------------------------------------------------
// Bidirectional LSTM for MI455X (gfx1250, wave32, WMMA, async-to-LDS).
//   - bf16 WMMA (v_wmma_f32_16x16x32_bf16) with fp32 accumulation.
//   - GEMMs: 64(M) x 128(N) block tile, A panels double-buffered in LDS via
//     global_load_async_to_lds_b128 (ASYNCcnt overlaps copies with WMMA);
//     inner loop software-pipelined (next kt's A/B fragments prefetched into
//     registers while current kt's WMMAs execute -> no per-wmma dscnt stall).
//   - Persistent recurrent kernel: 256 WGs, 256KB Wh slice pinned in LDS in
//     WMMA-B lane layout, h_prev async-staged per step (320KB LDS per WGP).
//   - Grid-wide step barrier: monotonic global atomic + s_sleep spin.
// ---------------------------------------------------------------------------

#define USE_ASYNC_COPY 1   // set 0 to fall back to plain load + ds_store

typedef __attribute__((ext_vector_type(16))) __bf16 v16bf;
typedef __attribute__((ext_vector_type(8)))  float  v8f;
typedef __attribute__((ext_vector_type(4)))  int    v4i;
typedef __attribute__((ext_vector_type(8)))  int    v8i;

static constexpr int Bc   = 16;
static constexpr int Sc   = 1024;
static constexpr int Dc   = 2048;
static constexpr int G4   = 4 * Dc;      // 8192
static constexpr int Mr   = Bc * Sc;     // 16384 rows (b*S+s)
static constexpr int KT_D = Dc / 32;     // 64 K-tiles for K=2048
static constexpr int NWGD = 128;         // workgroups per direction
static constexpr int NWG  = 2 * NWGD;    // persistent grid size
static constexpr int PKT  = 8;           // K-tiles per staged GEMM panel

static __device__ __forceinline__ unsigned short f2bf(float f) {
  union { float f; unsigned u; } x; x.f = f;
  unsigned u = x.u;
  u += 0x7FFFu + ((u >> 16) & 1u);       // round-to-nearest-even
  return (unsigned short)(u >> 16);
}

// 16B global -> LDS async copy. lds_off = byte offset into workgroup LDS
// (low 32 bits of the flat shared address == LDS offset per CDNA5 aperture).
static __device__ __forceinline__ void async_g2l16(unsigned lds_off,
                                                   const void* sbase, int voff) {
#if USE_ASYNC_COPY
  asm volatile("global_load_async_to_lds_b128 %0, %1, %2"
               :: "v"(lds_off), "v"(voff), "s"(sbase) : "memory");
#endif
}
// Device-scope variant: bypass/refresh WGP$ (source written by other WGPs).
static __device__ __forceinline__ void async_g2l16_dev(unsigned lds_off,
                                                       const void* sbase, int voff) {
#if USE_ASYNC_COPY
  asm volatile("global_load_async_to_lds_b128 %0, %1, %2 scope:SCOPE_DEV"
               :: "v"(lds_off), "v"(voff), "s"(sbase) : "memory");
#endif
}
static __device__ __forceinline__ void async_wait0() {
#if USE_ASYNC_COPY
  asm volatile("s_wait_asynccnt 0x0" ::: "memory");
#endif
}

// ---------------------------------------------------------------------------
__global__ void k_init(unsigned* __restrict__ cnt, unsigned short* __restrict__ hbuf) {
  int t = threadIdx.x + blockIdx.x * blockDim.x;
  if (t < 64) cnt[t] = 0u;
  for (int i = t; i < 2 * 2 * Bc * Dc; i += blockDim.x * gridDim.x) hbuf[i] = 0;
}

__global__ void k_f32_to_bf16(const float* __restrict__ src,
                              unsigned short* __restrict__ dst, int n) {
  int i = threadIdx.x + blockIdx.x * blockDim.x;
  int stride = blockDim.x * gridDim.x;
  for (; i < n; i += stride) dst[i] = f2bf(src[i]);
}

// ---------------------------------------------------------------------------
// Pack row-major W[K][N] fp32 into WMMA-B per-lane 32B records (bf16):
// record r: nt=r/(KT*32), kt=(r/32)%KT, lane=r%32;
//   n = nt*16 + lane%16 ; kbase = kt*32 + (lane/16)*16 ; 16 consecutive K.
// ---------------------------------------------------------------------------
__global__ void k_pack_b(const float* __restrict__ W,
                         unsigned short* __restrict__ P, int K, int N) {
  const int KT = K / 32;
  long r = threadIdx.x + (long)blockIdx.x * blockDim.x;
  long nrec = (long)(N / 16) * KT * 32;
  if (r >= nrec) return;
  int lane = (int)(r & 31);
  int kt   = (int)((r >> 5) % KT);
  int nt   = (int)(r / (32L * KT));
  int n    = nt * 16 + (lane & 15);
  int kb   = kt * 32 + (lane >> 4) * 16;
  unsigned short* out = P + r * 16;
#pragma unroll
  for (int j = 0; j < 16; ++j) out[j] = f2bf(W[(long)(kb + j) * N + n]);
}

// ---------------------------------------------------------------------------
// C[M,N] f32 = A[M,K] bf16 (row-major) x Bpacked (+ bias[n]).
// Block = 256 threads = 8 waves; block tile = 64(M) x 128(N).
// Wave w owns ntile blockIdx.x*8+w with 4 M-subtiles (B tile reused 4x).
// A staged per 8-ktile panel into LDS (WMMA-A lane layout) with async copies,
// double-buffered; inner kt loop register-pipelined one kt ahead.
// Dynamic LDS: 2 * PKT * 4 * 32 * 32B = 64KB.
// ---------------------------------------------------------------------------
__global__ void __launch_bounds__(256)
k_wmma_gemm(const unsigned short* __restrict__ A,
            const unsigned short* __restrict__ Bp,
            const float* __restrict__ bias, float* __restrict__ C,
            int N, int K) {
  const int KT   = K / 32;
  const int NP   = KT / PKT;
  const int lane = threadIdx.x & 31;
  const int w    = threadIdx.x >> 5;
  const int nt   = blockIdx.x * 8 + w;
  const int m0   = blockIdx.y * 64;
  const int nc   = nt * 16 + (lane & 15);

  extern __shared__ char smem[];
  const unsigned ldsBase = (unsigned)(size_t)(void*)smem;

  v8f acc[4];
  {
    float bv = bias ? bias[nc] : 0.f;
#pragma unroll
    for (int ms = 0; ms < 4; ++ms)
#pragma unroll
      for (int v = 0; v < 8; ++v) acc[ms][v] = bv;
  }

  // stage panel p (8 ktiles x 4 msubtiles) into buffer b (async)
  auto stage = [&](int p, int b) {
    const int ktbase = p * PKT;
    for (int r = threadIdx.x; r < PKT * 4 * 32; r += 256) {
      int lr   = r & 31;
      int msub = (r >> 5) & 3;
      int ktp  = r >> 7;
      int row  = m0 + msub * 16 + (lr & 15);
      int kel  = (ktbase + ktp) * 32 + (lr >> 4) * 16;
      int gb   = (int)(((long)row * K + kel) * 2);          // byte offset
      unsigned lo = ldsBase + (unsigned)((((b * PKT + ktp) * 4 + msub) * 32 + lr) * 32);
#if USE_ASYNC_COPY
      async_g2l16(lo, A, gb);
      async_g2l16(lo + 16, A, gb + 16);
#else
      const unsigned short* pA = A + (long)row * K + kel;
      *(v4i*)(smem + (size_t)(lo - ldsBase))      = *(const v4i*)pA;
      *(v4i*)(smem + (size_t)(lo - ldsBase) + 16) = *(const v4i*)(pA + 8);
#endif
    }
  };

  const unsigned short* brec = Bp + (long)nt * KT * 512 + lane * 16;
  auto ldB = [&](int kt) -> v16bf {
    return __builtin_bit_cast(v16bf, *(const v8i*)(brec + (long)kt * 512));
  };
  auto ldA = [&](int b, int ktp, int ms) -> v16bf {
    return *(const v16bf*)(smem +
        (size_t)((((b * PKT + ktp) * 4 + ms) * 32 + lane) * 32));
  };

  stage(0, 0);
  async_wait0();
  __syncthreads();

  int buf = 0;
  for (int p = 0; p < NP; ++p) {
    if (p + 1 < NP) stage(p + 1, buf ^ 1);   // async: overlaps compute below

    // register-pipelined: prefetch kt+1 fragments during kt's WMMAs
    v16bf bCur = ldB(p * PKT);
    v16bf aCur[4];
#pragma unroll
    for (int ms = 0; ms < 4; ++ms) aCur[ms] = ldA(buf, 0, ms);

#pragma unroll
    for (int ktp = 0; ktp < PKT; ++ktp) {
      v16bf bNxt;
      v16bf aNxt[4];
      if (ktp + 1 < PKT) {
        bNxt = ldB(p * PKT + ktp + 1);
#pragma unroll
        for (int ms = 0; ms < 4; ++ms) aNxt[ms] = ldA(buf, ktp + 1, ms);
      }
#pragma unroll
      for (int ms = 0; ms < 4; ++ms)
        acc[ms] = __builtin_amdgcn_wmma_f32_16x16x32_bf16(
            false, aCur[ms], false, bCur, (short)0, acc[ms], false, false);
      if (ktp + 1 < PKT) {
        bCur = bNxt;
#pragma unroll
        for (int ms = 0; ms < 4; ++ms) aCur[ms] = aNxt[ms];
      }
    }
    async_wait0();
    __syncthreads();
    buf ^= 1;
  }

#pragma unroll
  for (int ms = 0; ms < 4; ++ms)
#pragma unroll
    for (int v = 0; v < 8; ++v) {
      int m = m0 + ms * 16 + v + 8 * (lane >> 4);
      C[(long)m * N + nc] = acc[ms][v];
    }
}

// ---------------------------------------------------------------------------
// Persistent recurrent kernel. Grid = 256 WGs x 256 threads (8 waves).
// WG (dir, wg) owns hidden cols [wg*16, wg*16+16): Wh slice 2048x64 bf16
// = 256KB pinned in LDS (WMMA-B lane layout); h_prev async-staged to 64KB
// LDS (WMMA-A lane layout) each step; 8KB gate-exchange aliases A region.
// Wave (g, khalf) accumulates gate g over half of K; inner loop register-
// pipelined one kt ahead to keep the serial step latency minimal.
// ---------------------------------------------------------------------------
__global__ void __launch_bounds__(256)
k_lstm_rec(const unsigned short* __restrict__ WhP,   // packed, both dirs
           const float* __restrict__ xw,             // fp32 xW, both dirs
           const float* __restrict__ b_f, const float* __restrict__ b_b,
           unsigned short* __restrict__ hbuf,        // [dir][2][16][2048] bf16
           unsigned short* __restrict__ hcat,        // [16384][4096] bf16
           unsigned* __restrict__ cnt) {
  const int dir  = blockIdx.x / NWGD;
  const int wg   = blockIdx.x % NWGD;
  const int tid  = threadIdx.x;
  const int lane = tid & 31;
  const int w    = tid >> 5;     // 0..7
  const int g    = w & 3;        // gate (i,f,g,o)
  const int khlf = w >> 2;       // K-split half

  extern __shared__ char lds[];
  char*  ldsA = lds + 256 * 1024;             // 64*32 recs * 32B  =  64KB
  float* ldsZ = (float*)(lds + 256 * 1024);   // alias of ldsA: 8*256 f32
  const unsigned ldsBase  = (unsigned)(size_t)(void*)lds;
  const unsigned ldsABase = ldsBase + 256 * 1024;

  const unsigned short* Whp = WhP + (size_t)dir * ((size_t)512 * KT_D * 32 * 16);
  const float* xwD = xw + (size_t)dir * ((size_t)Mr * G4);
  const float* bias = dir ? b_b : b_f;

  // ---- one-time: async-preload this WG's Wh slice (4 gate ntiles) ----
  for (int r = tid; r < 4 * KT_D * 32; r += 256) {
    int lr = r & 31;
    int kt = (r >> 5) % KT_D;
    int gt = r / (32 * KT_D);
    int ntile = gt * 128 + wg;
    int gb = (int)((((long)ntile * KT_D + kt) * 32 + lr) * 32);  // bytes
#if USE_ASYNC_COPY
    async_g2l16(ldsBase + (unsigned)(r * 32), Whp, gb);
    async_g2l16(ldsBase + (unsigned)(r * 32) + 16, Whp, gb + 16);
#else
    *(v8i*)(lds + (size_t)r * 32) = *(const v8i*)((const char*)Whp + gb);
#endif
  }

  // gate-phase element for this thread: e = tid -> (m, j)
  const int m_e = tid >> 4;
  const int j_e = tid & 15;
  float c_st = 0.f;                      // fp32 cell state across all steps
  float bi  = bias[0 * Dc + wg * 16 + j_e];
  float bff = bias[1 * Dc + wg * 16 + j_e];
  float bgg = bias[2 * Dc + wg * 16 + j_e];
  float boo = bias[3 * Dc + wg * 16 + j_e];

  async_wait0();
  __syncthreads();

  auto ldA = [&](int kt) -> v16bf {
    return *(const v16bf*)(lds + 256 * 1024 + ((size_t)kt * 32 + lane) * 32);
  };
  auto ldB = [&](int kt) -> v16bf {
    return *(const v16bf*)(lds + (((size_t)g * KT_D + kt) * 32 + lane) * 32);
  };

  int cur = 0;
  unsigned short* hb = hbuf + (size_t)dir * (2L * Bc * Dc);
  const int ncol = g * Dc + wg * 16 + (lane & 15);

  for (int s = 0; s < Sc; ++s) {
    const int t_idx = dir ? (Sc - 1 - s) : s;

    // ---- async-stage h_prev -> ldsA in WMMA-A per-lane layout ----
    const unsigned short* hsrc = hb + (size_t)cur * (Bc * Dc);
    for (int r = tid; r < KT_D * 32; r += 256) {
      int lr = r & 31;
      int kt = r >> 5;
      int m  = lr & 15;
      int kh = lr >> 4;
      int gb = (int)(((long)m * Dc + kt * 32 + kh * 16) * 2);   // bytes
#if USE_ASYNC_COPY
      async_g2l16_dev(ldsABase + (unsigned)(r * 32), hsrc, gb);
      async_g2l16_dev(ldsABase + (unsigned)(r * 32) + 16, hsrc, gb + 16);
#else
      const unsigned short* p = hsrc + (size_t)m * Dc + kt * 32 + kh * 16;
      *(v4i*)(ldsA + (size_t)r * 32)      = *(const v4i*)p;
      *(v4i*)(ldsA + (size_t)r * 32 + 16) = *(const v4i*)(p + 8);
#endif
    }

    // ---- accumulator starts from fp32 xW_t (khalf 0 only) ----
    v8f acc;
    if (khlf == 0) {
#pragma unroll
      for (int v = 0; v < 8; ++v) {
        int m = v + 8 * (lane >> 4);
        acc[v] = xwD[((long)m * Sc + t_idx) * G4 + ncol];
      }
    } else {
#pragma unroll
      for (int v = 0; v < 8; ++v) acc[v] = 0.f;
    }

    async_wait0();
    __syncthreads();

    // ---- 32 chained WMMAs, register-pipelined one kt ahead ----
    const int kt0 = khlf * 32;
    v16bf aC = ldA(kt0);
    v16bf bC = ldB(kt0);
#pragma unroll 8
    for (int kt = kt0; kt < kt0 + 31; ++kt) {
      v16bf aN = ldA(kt + 1);
      v16bf bN = ldB(kt + 1);
      acc = __builtin_amdgcn_wmma_f32_16x16x32_bf16(false, aC, false, bC,
                                                    (short)0, acc, false, false);
      aC = aN; bC = bN;
    }
    acc = __builtin_amdgcn_wmma_f32_16x16x32_bf16(false, aC, false, bC,
                                                  (short)0, acc, false, false);

    // prefetch next step's xW_t bytes while GEMM results drain
    if (s + 1 < Sc) {
      int tn = dir ? (Sc - 2 - s) : (s + 1);
      __builtin_prefetch(&xwD[((long)(lane >> 4) * Sc + tn) * G4 + ncol], 0, 1);
    }

    __syncthreads();   // all waves done reading ldsA before ldsZ (alias) writes

#pragma unroll
    for (int v = 0; v < 8; ++v) {
      int m = v + 8 * (lane >> 4);
      ldsZ[w * 256 + m * 16 + (lane & 15)] = acc[v];
    }
    __syncthreads();

    // ---- gates: i,f,g,o -> c,h (one element per thread) ----
    {
      int e = tid;
      float zi = ldsZ[0 * 256 + e] + ldsZ[4 * 256 + e] + bi;
      float zf = ldsZ[1 * 256 + e] + ldsZ[5 * 256 + e] + bff;
      float zg = ldsZ[2 * 256 + e] + ldsZ[6 * 256 + e] + bgg;
      float zo = ldsZ[3 * 256 + e] + ldsZ[7 * 256 + e] + boo;
      float ig = 1.f / (1.f + __expf(-zi));
      float fg = 1.f / (1.f + __expf(-zf));
      float gg = tanhf(zg);
      float og = 1.f / (1.f + __expf(-zo));
      c_st = fg * c_st + ig * gg;
      float hv = og * tanhf(c_st);
      unsigned short hbf16 = f2bf(hv);
      int colD = wg * 16 + j_e;
      hb[(size_t)(cur ^ 1) * (Bc * Dc) + (size_t)m_e * Dc + colD] = hbf16;
      hcat[((size_t)m_e * Sc + t_idx) * (2 * Dc) + (size_t)dir * Dc + colD] = hbf16;
    }

    // ---- grid-wide step barrier (monotonic counter, s_sleep spin) ----
    __threadfence();
    __syncthreads();
    if (tid == 0) {
      __hip_atomic_fetch_add(cnt, 1u, __ATOMIC_ACQ_REL, __HIP_MEMORY_SCOPE_AGENT);
      unsigned target = (unsigned)(s + 1) * (unsigned)NWG;
      while (__hip_atomic_load(cnt, __ATOMIC_ACQUIRE, __HIP_MEMORY_SCOPE_AGENT) < target)
        __builtin_amdgcn_s_sleep(1);
    }
    __syncthreads();
    cur ^= 1;
  }
}

// ---------------------------------------------------------------------------
extern "C" void kernel_launch(void* const* d_in, const int* in_sizes, int n_in,
                              void* d_out, int out_size, void* d_ws, size_t ws_size,
                              hipStream_t stream) {
  (void)in_sizes; (void)n_in; (void)out_size; (void)ws_size;
  const float* x   = (const float*)d_in[0];
  const float* Wxf = (const float*)d_in[1];
  const float* Whf = (const float*)d_in[2];
  const float* bf  = (const float*)d_in[3];
  const float* Wxb = (const float*)d_in[4];
  const float* Whb = (const float*)d_in[5];
  const float* bb  = (const float*)d_in[6];
  const float* Wd  = (const float*)d_in[7];
  const float* bd  = (const float*)d_in[8];

  char* ws = (char*)d_ws;
  size_t off = 0;
  auto take = [&](size_t bytes) {
    size_t r = off; off += (bytes + 255) & ~(size_t)255; return r;
  };

  const long recWx = 512L * KT_D * 32;           // records per Wx/Wh matrix
  const long recWd = 128L * 128 * 32;            // records for Wd (K=4096,N=2048)

  unsigned*       cnt  = (unsigned*)(ws + take(256));
  unsigned short* xbf  = (unsigned short*)(ws + take((size_t)Mr * Dc * 2));
  unsigned short* WxP  = (unsigned short*)(ws + take((size_t)2 * recWx * 32));
  unsigned short* WhP  = (unsigned short*)(ws + take((size_t)2 * recWx * 32));
  unsigned short* WdP  = (unsigned short*)(ws + take((size_t)recWd * 32));
  float*          xw   = (float*)(ws + take((size_t)2 * Mr * G4 * 4));
  unsigned short* hcat = (unsigned short*)(ws + take((size_t)Mr * 2 * Dc * 2));
  unsigned short* hbuf = (unsigned short*)(ws + take((size_t)2 * 2 * Bc * Dc * 2));

  // state init (inside capture: deterministic across graph replays)
  k_init<<<dim3(64), dim3(256), 0, stream>>>(cnt, hbuf);
  k_f32_to_bf16<<<dim3(4096), dim3(256), 0, stream>>>(x, xbf, Mr * Dc);

  // pre-swizzle weights into WMMA-B lane layout (bf16)
  k_pack_b<<<dim3((unsigned)(recWx / 256)), dim3(256), 0, stream>>>(Wxf, WxP, Dc, G4);
  k_pack_b<<<dim3((unsigned)(recWx / 256)), dim3(256), 0, stream>>>(Wxb, WxP + recWx * 16, Dc, G4);
  k_pack_b<<<dim3((unsigned)(recWx / 256)), dim3(256), 0, stream>>>(Whf, WhP, Dc, G4);
  k_pack_b<<<dim3((unsigned)(recWx / 256)), dim3(256), 0, stream>>>(Whb, WhP + recWx * 16, Dc, G4);
  k_pack_b<<<dim3((unsigned)(recWd / 256)), dim3(256), 0, stream>>>(Wd, WdP, 2 * Dc, Dc);

  // hoisted input projections xW = x @ Wx (both directions), fp32 out
  const int gemmLds = 64 * 1024;
  hipFuncSetAttribute(reinterpret_cast<const void*>(k_wmma_gemm),
                      hipFuncAttributeMaxDynamicSharedMemorySize, gemmLds);
  k_wmma_gemm<<<dim3(G4 / 128, Mr / 64), dim3(256), gemmLds, stream>>>(
      xbf, WxP, nullptr, xw, G4, Dc);
  k_wmma_gemm<<<dim3(G4 / 128, Mr / 64), dim3(256), gemmLds, stream>>>(
      xbf, WxP + recWx * 16, nullptr, xw + (size_t)Mr * G4, G4, Dc);

  // persistent bidirectional recurrence (320KB dynamic LDS per WG)
  {
    const int ldsBytes = 320 * 1024;
    hipFuncSetAttribute(reinterpret_cast<const void*>(k_lstm_rec),
                        hipFuncAttributeMaxDynamicSharedMemorySize, ldsBytes);
    k_lstm_rec<<<dim3(NWG), dim3(256), ldsBytes, stream>>>(
        WhP, xw, bf, bb, hbuf, hcat, cnt);
  }

  // final dense: out = hcat @ Wd + bd  ->  d_out [B,S,D] fp32
  k_wmma_gemm<<<dim3(Dc / 128, Mr / 64), dim3(256), gemmLds, stream>>>(
      hcat, WdP, bd, (float*)d_out, Dc, 2 * Dc);
}
